// SSMDownstreamDecoder_27487790694541
// MI455X (gfx1250) — matrix operand order repaired and checked by reading, not compiled, not run on hardware
//
#include <hip/hip_runtime.h>
#include <hip/hip_bf16.h>

// ---------------- problem constants ----------------
#define SEQ     8192
#define HDIM    512
#define PDIM    256
#define NLAYERS 6
#define OUTD    128
#define NCHUNK  64
#define CHLEN   128

typedef __attribute__((ext_vector_type(16))) _Float16 v16h;
typedef __attribute__((ext_vector_type(8)))  float    v8f;

// ---------------- workspace layout (bytes), total ~= 77 MB ----------------
static constexpr size_t KB = 1024, MB = 1024 * 1024;
static constexpr size_t OFF_H      = 0;                      // 16MB  h (f32)
static constexpr size_t OFF_BU     = 16 * MB;                // 16MB  Bu / local xs (f32, cols [0,256)=re [256,512)=im)
static constexpr size_t OFF_ZH     = 32 * MB;                // 8MB   z  (f16)
static constexpr size_t OFF_XSH    = 40 * MB;                // 8MB   xs (f16, stacked re|im)
static constexpr size_t OFF_GH     = 48 * MB;                // 8MB   g  (f16)
static constexpr size_t OFF_HH     = 56 * MB;                // 8MB   x_h (encoder in), then h_h (f16)
static constexpr size_t OFF_WENC   = 64 * MB;                // 512KB padded W_enc^T f16 (K x N = 512x512)
static constexpr size_t OFF_WDEC   = OFF_WENC + 512 * KB;    // 128KB W_dec^T f16 (512x128)
static constexpr size_t OFF_BBAR   = OFF_WDEC + 128 * KB;    // 3MB   B-bar stacked^T f16 (6 x 512x512)
static constexpr size_t OFF_CST    = OFF_BBAR + 3 * MB;      // 3MB   C stacked^T f16
static constexpr size_t OFF_W1H    = OFF_CST  + 3 * MB;      // 3MB   W1^T f16
static constexpr size_t OFF_W2H    = OFF_W1H  + 3 * MB;      // 3MB   W2^T f16
static constexpr size_t OFF_BPAD   = OFF_W2H  + 3 * MB;      // 2KB   encoder bias (b_enc | ctx)
static constexpr size_t OFF_TAB    = OFF_BPAD + 4 * KB;      // 48KB  per-layer scalar tables (8 x 256 f32)
static constexpr size_t OFF_CARRY  = OFF_TAB  + 64 * KB;     // 128KB chunk carries (64 x 256 complex)
static constexpr size_t OFF_CSTATE = OFF_CARRY + 128 * KB;   // 128KB chunk entry states

// table arrays (256 floats each, per layer):
// 0:zre=Re(lam)*dt 1:zim 2:lambar_re 3:lambar_im 4:lambar^128 re 5:im 6:f_re 7:f_im  (f=(lambar-1)/lam)

// ---------------- WMMA fragment helpers (CDNA5 16x16x32 f16) ----------------
// A 16x32 f16 (row-major, ld = K): lane&15 = M row; lane>>4 adds K offset +8;
// halves cover K chunks [0..7] and [16..23] -> two contiguous b128 loads.
__device__ __forceinline__ v16h load_a_frag(const _Float16* __restrict__ A, int lda, int lane) {
    const _Float16* r = A + (lane & 15) * lda + ((lane >> 4) << 3);
    v16h a;
#pragma unroll
    for (int v = 0; v < 8; ++v) {
        int kk = ((v >> 2) << 4) + ((v & 3) << 1);
        a[2 * v]     = r[kk];
        a[2 * v + 1] = r[kk + 1];
    }
    return a;
}

// B 32x16 f16 from K-major weight WT (K x N row-major): lane&15 = n, lane>>4 adds K offset +8;
// element s -> k = 16*(s>>3) + (s&7), strided by ld.
__device__ __forceinline__ v16h load_b_frag(const _Float16* __restrict__ WTrow, int ld) {
    v16h b;
#pragma unroll
    for (int h = 0; h < 16; ++h) b[h] = WTrow[(size_t)(((h >> 3) << 4) + (h & 7)) * ld];
    return b;
}

__device__ __forceinline__ float gelu_tanh(float x) {
    float x3 = x * x * x;
    return 0.5f * x * (1.f + tanhf(0.7978845608028654f * (x + 0.044715f * x3)));
}
__device__ __forceinline__ float sigmoidf(float x) { return 1.f / (1.f + expf(-x)); }

// ---------------- GEMM kernels ----------------
// Block = 256 threads = 8 waves, one 16-row M strip per block.
// Each wave register-blocks TPW consecutive 16-col N tiles: one A-frag load
// feeds TPW WMMAs. Weights are pre-transposed f16 (K x N).

template <int TPW>
__global__ void __launch_bounds__(256) gemm_bias_kernel(const _Float16* __restrict__ A,
                                                        const _Float16* __restrict__ BT,
                                                        const float* __restrict__ bias,
                                                        float* __restrict__ C, int N, int K) {
    int lane = threadIdx.x & 31, wave = threadIdx.x >> 5;
    int mBase = blockIdx.x << 4;
    const _Float16* a0 = A + (size_t)mBase * K;
    int nTiles = N >> 4;
    for (int nt0 = wave * TPW; nt0 < nTiles; nt0 += 8 * TPW) {
        int nBase0 = nt0 << 4;
        v8f acc[TPW];
#pragma unroll
        for (int t = 0; t < TPW; ++t) acc[t] = {};
        const _Float16* b0 = BT + (size_t)((lane >> 4) << 3) * N + nBase0 + (lane & 15);
        for (int k0 = 0; k0 < K; k0 += 32) {
            v16h a = load_a_frag(a0 + k0, K, lane);
            const _Float16* br = b0 + (size_t)k0 * N;
#pragma unroll
            for (int t = 0; t < TPW; ++t) {
                v16h b = load_b_frag(br + t * 16, N);
                acc[t] = __builtin_amdgcn_wmma_f32_16x16x32_f16(false, a, false, b, (short)0,
                                                                acc[t], false, false);
            }
        }
        int m0 = mBase + ((lane >> 4) << 3);
#pragma unroll
        for (int t = 0; t < TPW; ++t) {
            int n = nBase0 + t * 16 + (lane & 15);
            float bv = bias ? bias[n] : 0.f;
#pragma unroll
            for (int r = 0; r < 8; ++r) C[(size_t)(m0 + r) * N + n] = acc[t][r] + bv;
        }
    }
}

// y = 2*acc + D[n]*z ; g = gelu(y)   (N = K = 512, TPW = 4)
__global__ void __launch_bounds__(256) gemm_y_gelu_kernel(const _Float16* __restrict__ xs,
                                                          const _Float16* __restrict__ CT,
                                                          const float* __restrict__ Dv,
                                                          const _Float16* __restrict__ zh,
                                                          _Float16* __restrict__ g) {
    constexpr int TPW = 4;
    int lane = threadIdx.x & 31, wave = threadIdx.x >> 5;
    int mBase = blockIdx.x << 4;
    const _Float16* a0 = xs + (size_t)mBase * HDIM;
    int nBase0 = (wave * TPW) << 4;
    v8f acc[TPW];
#pragma unroll
    for (int t = 0; t < TPW; ++t) acc[t] = {};
    const _Float16* b0 = CT + (size_t)((lane >> 4) << 3) * HDIM + nBase0 + (lane & 15);
    for (int k0 = 0; k0 < HDIM; k0 += 32) {
        v16h a = load_a_frag(a0 + k0, HDIM, lane);
        const _Float16* br = b0 + (size_t)k0 * HDIM;
#pragma unroll
        for (int t = 0; t < TPW; ++t) {
            v16h b = load_b_frag(br + t * 16, HDIM);
            acc[t] = __builtin_amdgcn_wmma_f32_16x16x32_f16(false, a, false, b, (short)0, acc[t],
                                                            false, false);
        }
    }
    int m0 = mBase + ((lane >> 4) << 3);
#pragma unroll
    for (int t = 0; t < TPW; ++t) {
        int n = nBase0 + t * 16 + (lane & 15);
        float dv = Dv[n];
#pragma unroll
        for (int r = 0; r < 8; ++r) {
            size_t idx = (size_t)(m0 + r) * HDIM + n;
            float y = 2.f * acc[t][r] + dv * (float)zh[idx];
            g[idx] = (_Float16)gelu_tanh(y);
        }
    }
}

// u1 = g@W1^T+b1 ; u2 = g@W2^T+b2 ; h += u1*sigmoid(u2) ; h_h = (f16)h   (TPW = 2)
__global__ void __launch_bounds__(256) gemm_glu_kernel(const _Float16* __restrict__ g,
                                                       const _Float16* __restrict__ W1T,
                                                       const _Float16* __restrict__ W2T,
                                                       const float* __restrict__ b1,
                                                       const float* __restrict__ b2,
                                                       float* __restrict__ h,
                                                       _Float16* __restrict__ hh) {
    constexpr int TPW = 2;
    int lane = threadIdx.x & 31, wave = threadIdx.x >> 5;
    int mBase = blockIdx.x << 4;
    const _Float16* a0 = g + (size_t)mBase * HDIM;
    for (int nt0 = wave * TPW; nt0 < (HDIM >> 4); nt0 += 8 * TPW) {
        int nBase0 = nt0 << 4;
        v8f acc1[TPW], acc2[TPW];
#pragma unroll
        for (int t = 0; t < TPW; ++t) { acc1[t] = {}; acc2[t] = {}; }
        const _Float16* p1 = W1T + (size_t)((lane >> 4) << 3) * HDIM + nBase0 + (lane & 15);
        const _Float16* p2 = W2T + (size_t)((lane >> 4) << 3) * HDIM + nBase0 + (lane & 15);
        for (int k0 = 0; k0 < HDIM; k0 += 32) {
            v16h a = load_a_frag(a0 + k0, HDIM, lane);
            const _Float16* r1 = p1 + (size_t)k0 * HDIM;
            const _Float16* r2 = p2 + (size_t)k0 * HDIM;
#pragma unroll
            for (int t = 0; t < TPW; ++t) {
                v16h f1 = load_b_frag(r1 + t * 16, HDIM);
                v16h f2 = load_b_frag(r2 + t * 16, HDIM);
                acc1[t] = __builtin_amdgcn_wmma_f32_16x16x32_f16(false, a, false, f1, (short)0,
                                                                 acc1[t], false, false);
                acc2[t] = __builtin_amdgcn_wmma_f32_16x16x32_f16(false, a, false, f2, (short)0,
                                                                 acc2[t], false, false);
            }
        }
        int m0 = mBase + ((lane >> 4) << 3);
#pragma unroll
        for (int t = 0; t < TPW; ++t) {
            int n = nBase0 + t * 16 + (lane & 15);
            float bb1 = b1[n], bb2 = b2[n];
#pragma unroll
            for (int r = 0; r < 8; ++r) {
                size_t idx = (size_t)(m0 + r) * HDIM + n;
                float v  = (acc1[t][r] + bb1) * sigmoidf(acc2[t][r] + bb2);
                float nh = h[idx] + v;
                h[idx]  = nh;
                hh[idx] = (_Float16)nh;
            }
        }
    }
}

// ---------------- LayerNorm: one row (512) per block ----------------
__global__ void __launch_bounds__(256) ln_kernel(const float* __restrict__ h,
                                                 const float* __restrict__ scale,
                                                 const float* __restrict__ bias,
                                                 _Float16* __restrict__ z) {
    __shared__ float r1[256], r2[256];
    int row = blockIdx.x, t = threadIdx.x;
    const float* hr = h + (size_t)row * HDIM;
    float a = hr[t], b = hr[t + 256];
    r1[t] = a + b;
    r2[t] = a * a + b * b;
    __syncthreads();
    for (int s = 128; s > 0; s >>= 1) {
        if (t < s) { r1[t] += r1[t + s]; r2[t] += r2[t + s]; }
        __syncthreads();
    }
    float mu  = r1[0] * (1.f / HDIM);
    float var = r2[0] * (1.f / HDIM) - mu * mu;
    float inv = rsqrtf(var + 1e-5f);
    size_t base = (size_t)row * HDIM;
    z[base + t]       = (_Float16)((a - mu) * inv * scale[t] + bias[t]);
    z[base + t + 256] = (_Float16)((b - mu) * inv * scale[t + 256] + bias[t + 256]);
}

// ---------------- parameter prep ----------------
__global__ void prep_scalars(const float* __restrict__ Lre_, const float* __restrict__ Lim_,
                             const float* __restrict__ log_step, float* __restrict__ tab) {
    int layer = blockIdx.x, p = threadIdx.x;
    int i = layer * PDIM + p;
    float dt  = expf(log_step[i]);
    float Lre = Lre_[i], Lim = Lim_[i];
    float zre = Lre * dt, zim = Lim * dt;
    float s, c;
    float e = expf(zre);
    __sincosf(zim, &s, &c);
    float lbre = e * c, lbim = e * s;
    float e128 = expf(128.f * zre);
    float s2, c2;
    __sincosf(128.f * zim, &s2, &c2);
    float nr = lbre - 1.f, ni = lbim;
    float den = Lre * Lre + Lim * Lim;
    float* T = tab + (size_t)layer * 8 * PDIM;
    T[0 * PDIM + p] = zre;
    T[1 * PDIM + p] = zim;
    T[2 * PDIM + p] = lbre;
    T[3 * PDIM + p] = lbim;
    T[4 * PDIM + p] = e128 * c2;
    T[5 * PDIM + p] = e128 * s2;
    T[6 * PDIM + p] = (nr * Lre + ni * Lim) / den;
    T[7 * PDIM + p] = (ni * Lre - nr * Lim) / den;
}

// B-bar stacked, TRANSPOSED (K-major 512x512 per layer): dst[k*512 + n],
// n in [0,256)=Re-row p=n, n in [256,512)=Im-row p=n-256 ; f=(lambar-1)/lam
__global__ void prep_bbar(const float* __restrict__ Bre, const float* __restrict__ Bim,
                          const float* __restrict__ tab, _Float16* __restrict__ out) {
    int layer = blockIdx.y;
    int idx = blockIdx.x * blockDim.x + threadIdx.x;  // < 512*512
    int k = idx >> 9, n = idx & 511, p = n & 255;
    const float* T = tab + (size_t)layer * 8 * PDIM;
    float fre = T[6 * PDIM + p], fim = T[7 * PDIM + p];
    size_t bi = (size_t)layer * PDIM * HDIM + (size_t)p * HDIM + k;
    float br = Bre[bi], bm = Bim[bi];
    float v = (n < 256) ? (fre * br - fim * bm) : (fre * bm + fim * br);
    out[(size_t)layer * HDIM * HDIM + idx] = (_Float16)v;
}

// C stacked, TRANSPOSED (K-major 512x512 per layer): dst[kk*512 + n],
// kk<256 -> C_re[n,kk], kk>=256 -> -C_im[n,kk-256]
__global__ void prep_cstack(const float* __restrict__ Cre, const float* __restrict__ Cim,
                            _Float16* __restrict__ out) {
    int layer = blockIdx.y;
    int idx = blockIdx.x * blockDim.x + threadIdx.x;
    int kk = idx >> 9, n = idx & 511;
    size_t base = (size_t)layer * HDIM * PDIM + (size_t)n * PDIM;
    float v = (kk < 256) ? Cre[base + kk] : -Cim[base + (kk - 256)];
    out[(size_t)layer * HDIM * HDIM + idx] = (_Float16)v;
}

// generic NxK f32 -> K-major (KxN) f16 transpose-convert; blockIdx.y = matrix index
__global__ void convert_f16_T(const float* __restrict__ src, _Float16* __restrict__ dst,
                              int N, int K) {
    int total = N * K;
    int idx = blockIdx.x * blockDim.x + threadIdx.x;
    if (idx >= total) return;
    int k = idx / N, n = idx - k * N;
    size_t off = (size_t)blockIdx.y * total;
    dst[off + idx] = (_Float16)src[off + (size_t)n * K + k];
}

__global__ void convert_f16(const float* __restrict__ src, _Float16* __restrict__ dst, int n) {
    int i = blockIdx.x * blockDim.x + threadIdx.x;
    if (i < n) dst[i] = (_Float16)src[i];
}

// padded W_enc, TRANSPOSED: dst[k*512 + n] = (n < 504) ? W_enc[n,k] : 0
__global__ void prep_wencpad(const float* __restrict__ Wenc, _Float16* __restrict__ dst) {
    int idx = blockIdx.x * blockDim.x + threadIdx.x;  // < 512*512
    int k = idx >> 9, n = idx & 511;
    dst[idx] = (_Float16)((n < HDIM - 8) ? Wenc[(size_t)n * HDIM + k] : 0.f);
}

__global__ void prep_bpad(const float* __restrict__ benc, const float* __restrict__ ctx,
                          float* __restrict__ bpad) {
    int n = threadIdx.x;
    bpad[n] = (n < HDIM - 8) ? benc[n] : ctx[n - (HDIM - 8)];
}

// ---------------- chunked parallel scan ----------------
__global__ void __launch_bounds__(256) scan_pass1(float* __restrict__ Bu,
                                                  const float* __restrict__ tab,
                                                  const float* __restrict__ state0,
                                                  float* __restrict__ carry) {
    int p = threadIdx.x, chunk = blockIdx.x;
    float lbre = tab[2 * PDIM + p], lbim = tab[3 * PDIM + p];
    if (chunk == 0) {  // fold initial state: Bu[0] += lambar * x0
        float x0r = state0[p * 2], x0i = state0[p * 2 + 1];
        Bu[p]       += lbre * x0r - lbim * x0i;
        Bu[256 + p] += lbre * x0i + lbim * x0r;
    }
    float xr = 0.f, xi = 0.f;
    int t0 = chunk * CHLEN;
    for (int t = 0; t < CHLEN; ++t) {
        size_t idx = (size_t)(t0 + t) * HDIM + p;
        float br = Bu[idx], bi = Bu[idx + 256];
        float nr = lbre * xr - lbim * xi + br;
        float ni = lbre * xi + lbim * xr + bi;
        xr = nr; xi = ni;
        Bu[idx] = xr; Bu[idx + 256] = xi;
    }
    carry[(size_t)(chunk * PDIM + p) * 2]     = xr;
    carry[(size_t)(chunk * PDIM + p) * 2 + 1] = xi;
}

__global__ void __launch_bounds__(256) scan_carry(const float* __restrict__ carry,
                                                  float* __restrict__ cstate,
                                                  const float* __restrict__ tab) {
    int p = threadIdx.x;
    float ar = tab[4 * PDIM + p], ai = tab[5 * PDIM + p];  // lambar^128
    float Xr = 0.f, Xi = 0.f;
    for (int c = 0; c < NCHUNK; ++c) {
        cstate[(size_t)(c * PDIM + p) * 2]     = Xr;
        cstate[(size_t)(c * PDIM + p) * 2 + 1] = Xi;
        float cr = carry[(size_t)(c * PDIM + p) * 2];
        float ci = carry[(size_t)(c * PDIM + p) * 2 + 1];
        float nXr = ar * Xr - ai * Xi + cr;
        float nXi = ar * Xi + ai * Xr + ci;
        Xr = nXr; Xi = nXi;
    }
}

__global__ void __launch_bounds__(256) scan_fixup(const float* __restrict__ Bu,
                                                  const float* __restrict__ cstate,
                                                  const float* __restrict__ tab,
                                                  _Float16* __restrict__ xsh,
                                                  float* __restrict__ stateOut) {
    int t = blockIdx.x, p = threadIdx.x;
    int chunk = t >> 7, tl = t & (CHLEN - 1);
    float Xr = cstate[(size_t)(chunk * PDIM + p) * 2];
    float Xi = cstate[(size_t)(chunk * PDIM + p) * 2 + 1];
    float tt = (float)(tl + 1);
    float e = expf(tab[0 * PDIM + p] * tt);
    float s, c;
    __sincosf(tab[1 * PDIM + p] * tt, &s, &c);
    float pr = e * c, pi = e * s;
    size_t idx = (size_t)t * HDIM + p;
    float xr = Bu[idx]       + pr * Xr - pi * Xi;
    float xi = Bu[idx + 256] + pr * Xi + pi * Xr;
    xsh[idx]       = (_Float16)xr;
    xsh[idx + 256] = (_Float16)xi;
    if (t == SEQ - 1) { stateOut[p * 2] = xr; stateOut[p * 2 + 1] = xi; }
}

// ---------------- host orchestration ----------------
extern "C" void kernel_launch(void* const* d_in, const int* in_sizes, int n_in,
                              void* d_out, int out_size, void* d_ws, size_t ws_size,
                              hipStream_t stream) {
    const float* x        = (const float*)d_in[0];
    const float* state    = (const float*)d_in[1];
    const float* ctx      = (const float*)d_in[2];
    const float* W_enc    = (const float*)d_in[3];
    const float* b_enc    = (const float*)d_in[4];
    const float* Lam_re   = (const float*)d_in[5];
    const float* Lam_im   = (const float*)d_in[6];
    const float* B_re     = (const float*)d_in[7];
    const float* B_im     = (const float*)d_in[8];
    const float* C_re     = (const float*)d_in[9];
    const float* C_im     = (const float*)d_in[10];
    const float* Dv       = (const float*)d_in[11];
    const float* log_step = (const float*)d_in[12];
    const float* ln_scale = (const float*)d_in[13];
    const float* ln_bias  = (const float*)d_in[14];
    const float* W1       = (const float*)d_in[15];
    const float* b1       = (const float*)d_in[16];
    const float* W2       = (const float*)d_in[17];
    const float* b2       = (const float*)d_in[18];
    const float* W_dec    = (const float*)d_in[19];
    const float* b_dec    = (const float*)d_in[20];

    char* ws = (char*)d_ws;
    float*    h      = (float*)(ws + OFF_H);
    float*    Bu     = (float*)(ws + OFF_BU);
    _Float16* zh     = (_Float16*)(ws + OFF_ZH);
    _Float16* xsh    = (_Float16*)(ws + OFF_XSH);
    _Float16* gh     = (_Float16*)(ws + OFF_GH);
    _Float16* hh     = (_Float16*)(ws + OFF_HH);   // x_h first, then h_h
    _Float16* wencT  = (_Float16*)(ws + OFF_WENC);
    _Float16* wdecT  = (_Float16*)(ws + OFF_WDEC);
    _Float16* bbarT  = (_Float16*)(ws + OFF_BBAR);
    _Float16* cstT   = (_Float16*)(ws + OFF_CST);
    _Float16* w1T    = (_Float16*)(ws + OFF_W1H);
    _Float16* w2T    = (_Float16*)(ws + OFF_W2H);
    float*    bpad   = (float*)(ws + OFF_BPAD);
    float*    tab    = (float*)(ws + OFF_TAB);
    float*    carry  = (float*)(ws + OFF_CARRY);
    float*    cstate = (float*)(ws + OFF_CSTATE);

    float* y_out      = (float*)d_out;
    float* states_out = y_out + (size_t)SEQ * OUTD;

    // ---- parameter prep (discretization + f32 -> f16 transposed weights) ----
    prep_scalars<<<NLAYERS, PDIM, 0, stream>>>(Lam_re, Lam_im, log_step, tab);
    dim3 gL((HDIM * HDIM) / 256, NLAYERS);
    prep_bbar<<<gL, 256, 0, stream>>>(B_re, B_im, tab, bbarT);
    prep_cstack<<<gL, 256, 0, stream>>>(C_re, C_im, cstT);
    convert_f16_T<<<gL, 256, 0, stream>>>(W1, w1T, HDIM, HDIM);
    convert_f16_T<<<gL, 256, 0, stream>>>(W2, w2T, HDIM, HDIM);
    convert_f16_T<<<dim3((OUTD * HDIM + 255) / 256, 1), 256, 0, stream>>>(W_dec, wdecT, OUTD, HDIM);
    convert_f16<<<(SEQ * HDIM + 255) / 256, 256, 0, stream>>>(x, hh, SEQ * HDIM);  // x_h
    prep_wencpad<<<(HDIM * HDIM) / 256, 256, 0, stream>>>(W_enc, wencT);
    prep_bpad<<<1, HDIM, 0, stream>>>(b_enc, ctx, bpad);

    // ---- encoder: h = [x@W_enc^T + b_enc | ctx]  (pad trick) ----
    gemm_bias_kernel<4><<<SEQ / 16, 256, 0, stream>>>(hh, wencT, bpad, h, HDIM, HDIM);

    // ---- layers ----
    for (int l = 0; l < NLAYERS; ++l) {
        size_t wOff = (size_t)l * HDIM * HDIM;
        const float* T = tab + (size_t)l * 8 * PDIM;
        ln_kernel<<<SEQ, 256, 0, stream>>>(h, ln_scale + l * HDIM, ln_bias + l * HDIM, zh);
        gemm_bias_kernel<4><<<SEQ / 16, 256, 0, stream>>>(zh, bbarT + wOff, nullptr, Bu, HDIM,
                                                          HDIM);
        scan_pass1<<<NCHUNK, PDIM, 0, stream>>>(Bu, T, state + (size_t)l * PDIM * 2, carry);
        scan_carry<<<1, PDIM, 0, stream>>>(carry, cstate, T);
        scan_fixup<<<SEQ, PDIM, 0, stream>>>(Bu, cstate, T, xsh,
                                             states_out + (size_t)l * PDIM * 2);
        gemm_y_gelu_kernel<<<SEQ / 16, 256, 0, stream>>>(xsh, cstT + wOff, Dv + l * HDIM, zh, gh);
        gemm_glu_kernel<<<SEQ / 16, 256, 0, stream>>>(gh, w1T + wOff, w2T + wOff,
                                                      b1 + l * HDIM, b2 + l * HDIM, h, hh);
    }

    // ---- decoder: y = h @ W_dec^T + b_dec ----
    gemm_bias_kernel<1><<<SEQ / 16, 256, 0, stream>>>(hh, wdecT, b_dec, y_out, OUTD, HDIM);
}
